// RetrievalGumbelSoftmax_70849780515091
// MI455X (gfx1250) — matrix operand order repaired and checked by reading, not compiled, not run on hardware
//
#include <hip/hip_runtime.h>
#include <math.h>

// ---------------- problem constants ----------------
#define B_Q    1024
#define D_DIM  1024
#define N_DB   100000
#define TOPK   70
#define NPAD   100096          // 782 * 128 (N padded to tile multiple)
#define NBLK   782             // N tiles of 128
#define MBLK   8               // M tiles of 128
#define TK     32              // WMMA K per step (bf16)
#define LDSW   40              // LDS row stride in bf16 elems (32 + 8 pad = 80B)
#define CHUNK  2048            // top-k phase-1 chunk
#define NCH    49              // ceil(NPAD / CHUNK)
#define MERGE  4096            // >= NCH*TOPK = 3430, power of 2

typedef __attribute__((ext_vector_type(16))) __bf16        v16bf;
typedef __attribute__((ext_vector_type(8)))  float         v8f;
typedef __attribute__((ext_vector_type(4)))  unsigned int  u32x4;
typedef __attribute__((ext_vector_type(4)))  float         f32x4;
typedef __attribute__((ext_vector_type(4)))  unsigned int  tdm_v4u;
typedef __attribute__((ext_vector_type(4)))  int           tdm_v4i;
typedef __attribute__((ext_vector_type(8)))  int           tdm_v8i;

union Frag { v16bf v; u32x4 q[2]; };

__device__ __forceinline__ unsigned short f2bf(float f) {
  unsigned u = __float_as_uint(f);
  u += 0x7FFFu + ((u >> 16) & 1u);       // round-to-nearest-even
  return (unsigned short)(u >> 16);
}

// Issue one TDM 2-D tile load: rows x 32 bf16 elements, global row stride
// D_DIM elems, LDS rows padded to 80B via D# pad fields (16 DWORDs data +
// 4 DWORDs pad). OOB rows (y >= rows) read as zero.
__device__ __forceinline__ void tdm_load_tile(const unsigned short* gsrc,
                                              unsigned lds_addr, int rows) {
  unsigned long long ga = (unsigned long long)(size_t)gsrc;
  tdm_v4u g0;
  g0[0] = 1u;                                   // count=1 (valid user D#)
  g0[1] = lds_addr;                             // lds_addr -> bits[63:32]
  g0[2] = (unsigned)(ga & 0xFFFFFFFFu);         // global_addr[31:0]
  g0[3] = (unsigned)((ga >> 32) & 0x01FFFFFFu)  // global_addr[56:32]
        | (2u << 30);                           // type=2 ("image")
  tdm_v8i g1;
  g1[0] = (1 << 16)                             // data_size = 2 bytes
        | (1 << 20)                             // pad_enable
        | (3 << 22)                             // pad_interval: 16 DWORDs
        | (3 << 25);                            // pad_amount:   4 DWORDs
  g1[1] = (int)((D_DIM & 0xFFFF) << 16);        // tensor_dim0[15:0] = 1024
  g1[2] = (int)((D_DIM >> 16) & 0xFFFF)         // tensor_dim0[31:16]
        | (int)((rows & 0xFFFF) << 16);         // tensor_dim1[15:0] = rows
  g1[3] = (int)(TK << 16);                      // tile_dim0 = 32 elems
  g1[4] = 128;                                  // tile_dim1 = 128, tile_dim2 = 0
  g1[5] = D_DIM;                                // tensor_dim0_stride = 1024
  g1[6] = 0;
  g1[7] = 0;
  tdm_v4i g2 = {0, 0, 0, 0};                    // 2-D tensor: groups 2/3 unused
  tdm_v4i g3 = {0, 0, 0, 0};
#if __clang_major__ >= 23
  tdm_v8i g4 = {0, 0, 0, 0, 0, 0, 0, 0};
  __builtin_amdgcn_tensor_load_to_lds(g0, g1, g2, g3, g4, 0);
#else
  __builtin_amdgcn_tensor_load_to_lds(g0, g1, g2, g3, 0);
#endif
}

// ---------------- kernel 1: per-row inverse L2 norms (one wave32 / row) ----
__global__ void rownorm_kernel(const float* __restrict__ q,
                               const float* __restrict__ db,
                               float* __restrict__ qrinv,
                               float* __restrict__ dbrinv) {
  int wave = threadIdx.x >> 5, lane = threadIdx.x & 31;
  long row = (long)blockIdx.x * 8 + wave;
  if (row >= (long)(B_Q + N_DB)) return;
  const float* src = (row < B_Q) ? (q + row * D_DIM)
                                 : (db + (row - B_Q) * (long)D_DIM);
  const f32x4* p = (const f32x4*)src;
  float s = 0.f;
  #pragma unroll
  for (int c = lane; c < D_DIM / 4; c += 32) {
    f32x4 v = p[c];
    s += v[0]*v[0] + v[1]*v[1] + v[2]*v[2] + v[3]*v[3];
  }
  #pragma unroll
  for (int off = 16; off; off >>= 1) s += __shfl_xor(s, off, 32);
  if (lane == 0) {
    float inv = 1.0f / fmaxf(sqrtf(s), 1e-12f);   // F.normalize semantics
    if (row < B_Q) qrinv[row] = inv; else dbrinv[row - B_Q] = inv;
  }
}

// ---------------- kernel 2: normalize + convert to bf16 (one block / row) --
__global__ void tobf16_kernel(const float* __restrict__ q,
                              const float* __restrict__ db,
                              const float* __restrict__ qrinv,
                              const float* __restrict__ dbrinv,
                              unsigned short* __restrict__ qn,
                              unsigned short* __restrict__ dbn) {
  long row = blockIdx.x;
  int tid = threadIdx.x;
  const float* src; float inv; unsigned short* dst;
  if (row < B_Q) { src = q + row * D_DIM; inv = qrinv[row]; dst = qn + row * D_DIM; }
  else { long r = row - B_Q; src = db + r * (long)D_DIM; inv = dbrinv[r]; dst = dbn + r * (long)D_DIM; }
  f32x4 v = ((const f32x4*)src)[tid];
  unsigned lo = (unsigned)f2bf(v[0]*inv) | ((unsigned)f2bf(v[1]*inv) << 16);
  unsigned hi = (unsigned)f2bf(v[2]*inv) | ((unsigned)f2bf(v[3]*inv) << 16);
  ((unsigned int*)dst)[tid*2]   = lo;
  ((unsigned int*)dst)[tid*2+1] = hi;
}

// ---------------- kernel 3: WMMA GEMM sims = qn * dbn^T ----------------
// 128x128 tile / workgroup, 8 waves, each wave = 32(M) x 64(N) = 2x4 WMMA
// tiles. Double-buffered LDS: A staged by async global->LDS DMA (ASYNCcnt),
// B (the streaming 205MB operand) staged by the Tensor Data Mover
// (TENSORcnt), both overlapped with the WMMA pipe.
__global__ __launch_bounds__(256)
void gemm_sims_kernel(const unsigned short* __restrict__ qn,
                      const unsigned short* __restrict__ dbn,
                      float* __restrict__ sims) {
  __shared__ __attribute__((aligned(16))) unsigned short As[2][128 * LDSW];
  __shared__ __attribute__((aligned(16))) unsigned short Bs[2][128 * LDSW];

  const int tid  = threadIdx.x;
  const int lane = tid & 31;
  const int wave = tid >> 5;
  const int wm   = wave & 3;        // 4 waves along M (32 rows each)
  const int wn   = wave >> 2;       // 2 waves along N (64 cols each)
  const int mbase = blockIdx.y * 128;
  const int nbase = blockIdx.x * 128;
  const int brows = (N_DB - nbase < 128) ? (N_DB - nbase) : 128; // TDM OOB rows -> 0

  v8f acc[2][4];
  #pragma unroll
  for (int mt = 0; mt < 2; ++mt)
    #pragma unroll
    for (int nt = 0; nt < 4; ++nt)
      #pragma unroll
      for (int i = 0; i < 8; ++i) acc[mt][nt][i] = 0.0f;

  const int khalf = lane >> 4;      // which K-half this lane holds
  const int r16   = lane & 15;

  // stage one K-slab: A tile via per-thread async DMA chunks, B tile via TDM
  auto stage = [&](int buf, int kk) {
    #pragma unroll
    for (int s = 0; s < 2; ++s) {
      int c   = (tid << 1) | s;     // 0..511 chunks of 16B
      int row = c >> 2;             // 0..127
      int kc  = c & 3;              // 4 chunks of 8 bf16
      const unsigned short* ga = qn + (size_t)(mbase + row) * D_DIM + kk + kc * 8;
      unsigned la = (unsigned)(size_t)&As[buf][row * LDSW + kc * 8];
      asm volatile("global_load_async_to_lds_b128 %0, %1, off"
                   :: "v"(la), "v"(ga) : "memory");
    }
    if (wave == 0)
      tdm_load_tile(dbn + (size_t)nbase * D_DIM + kk,
                    (unsigned)(size_t)&Bs[buf][0], brows);
  };
  auto wait_stage = [&]() {
    asm volatile("s_wait_asynccnt 0x0" ::: "memory");
    if (wave == 0) __builtin_amdgcn_s_wait_tensorcnt(0);
    __syncthreads();
  };

  stage(0, 0);                                    // prologue fill of buffer 0
  wait_stage();

  for (int kk = 0; kk < D_DIM; kk += TK) {
    const int cur = (kk >> 5) & 1;
    if (kk + TK < D_DIM) stage(cur ^ 1, kk + TK); // DMA fill of other buffer
    if (kk + 2 * TK < D_DIM)                      // prefetch A 2 slabs ahead
      __builtin_prefetch(qn + (size_t)(mbase + (tid >> 1)) * D_DIM + kk + 2 * TK, 0, 1);

    // ---- load fragments per ISA 7.12.2 layouts, compute 8 WMMAs ----
    Frag fa[2], fb[4];
    #pragma unroll
    for (int mt = 0; mt < 2; ++mt) {
      // A 16x32 bf16: lane<16 -> K{0..7,16..23}, lane>=16 -> K{8..15,24..31}
      const unsigned short* p = &As[cur][(wm * 32 + mt * 16 + r16) * LDSW + khalf * 8];
      fa[mt].q[0] = *(const u32x4*)(p);
      fa[mt].q[1] = *(const u32x4*)(p + 16);
    }
    #pragma unroll
    for (int nt = 0; nt < 4; ++nt) {
      // B 32x16 bf16: lane holds column (lane&15), lane<16 -> K{0..7,16..23}, lane>=16 -> K{8..15,24..31}
      const unsigned short* p = &Bs[cur][(wn * 64 + nt * 16 + r16) * LDSW + khalf * 8];
      fb[nt].q[0] = *(const u32x4*)(p);
      fb[nt].q[1] = *(const u32x4*)(p + 16);
    }
    #pragma unroll
    for (int mt = 0; mt < 2; ++mt)
      #pragma unroll
      for (int nt = 0; nt < 4; ++nt)
        acc[mt][nt] = __builtin_amdgcn_wmma_f32_16x16x32_bf16(
            /*neg_a=*/false, fa[mt].v, /*neg_b=*/false, fb[nt].v,
            /*c_mod=*/(short)0, acc[mt][nt],
            /*reuse_a=*/false, /*reuse_b=*/false);

    // buffer cur^1 (written by this iteration's DMAs) is read only after
    // this wait + barrier, so both DMA engines overlap the WMMAs above.
    wait_stage();
  }

  // ---- store C: VGPR v -> M = v + 8*(lane>=16), N = lane&15 ----
  const int mrow0 = mbase + wm * 32;
  const int ncol0 = nbase + wn * 64;
  const int mhi   = khalf * 8;
  #pragma unroll
  for (int mt = 0; mt < 2; ++mt)
    #pragma unroll
    for (int nt = 0; nt < 4; ++nt) {
      int ncol = ncol0 + nt * 16 + r16;
      bool valid = ncol < N_DB;
      #pragma unroll
      for (int v = 0; v < 8; ++v) {
        int mrow = mrow0 + mt * 16 + mhi + v;
        sims[(size_t)mrow * NPAD + ncol] = valid ? acc[mt][nt][v] : -1e30f;
      }
    }
}

// ---------------- kernel 4: per-(row,chunk) top-70 via bitonic sort --------
__global__ void topk_chunk_kernel(const float* __restrict__ sims,
                                  float* __restrict__ candV,
                                  int* __restrict__ candI) {
  __shared__ float sv[CHUNK];
  __shared__ int   si[CHUNK];
  int b = blockIdx.y, ch = blockIdx.x, tid = threadIdx.x;
  int base = ch * CHUNK;
  for (int i = tid; i < CHUNK; i += 256) {
    int n = base + i;
    sv[i] = (n < NPAD) ? sims[(size_t)b * NPAD + n] : -3e30f;
    si[i] = n;
  }
  __syncthreads();
  for (int k = 2; k <= CHUNK; k <<= 1)
    for (int j = k >> 1; j > 0; j >>= 1) {
      for (int i = tid; i < CHUNK; i += 256) {
        int ixj = i ^ j;
        if (ixj > i) {
          bool desc = ((i & k) == 0);
          float a = sv[i], c = sv[ixj];
          if (desc ? (a < c) : (a > c)) {
            sv[i] = c; sv[ixj] = a;
            int t = si[i]; si[i] = si[ixj]; si[ixj] = t;
          }
        }
      }
      __syncthreads();
    }
  if (tid < TOPK) {
    size_t o = ((size_t)b * NCH + ch) * TOPK + tid;
    candV[o] = sv[tid]; candI[o] = si[tid];
  }
}

// ---------------- kernel 5: merge candidates + Gumbel softmax --------------
__global__ void merge_softmax_kernel(const float* __restrict__ candV,
                                     const int* __restrict__ candI,
                                     const float* __restrict__ noise,
                                     float* __restrict__ attn,
                                     int* __restrict__ topidx) {
  __shared__ float sv[MERGE];
  __shared__ int   si[MERGE];
  __shared__ float ex[TOPK];
  __shared__ float rdenom;
  int b = blockIdx.x, tid = threadIdx.x;
  const int tot = NCH * TOPK;       // 3430
  for (int i = tid; i < MERGE; i += 256) {
    if (i < tot) { sv[i] = candV[(size_t)b * tot + i]; si[i] = candI[(size_t)b * tot + i]; }
    else         { sv[i] = -3e30f;                     si[i] = 0; }
  }
  __syncthreads();
  for (int k = 2; k <= MERGE; k <<= 1)
    for (int j = k >> 1; j > 0; j >>= 1) {
      for (int i = tid; i < MERGE; i += 256) {
        int ixj = i ^ j;
        if (ixj > i) {
          bool desc = ((i & k) == 0);
          float a = sv[i], c = sv[ixj];
          if (desc ? (a < c) : (a > c)) {
            sv[i] = c; sv[ixj] = a;
            int t = si[i]; si[i] = si[ixj]; si[ixj] = t;
          }
        }
      }
      __syncthreads();
    }
  // sorted descending: slot k pairs with noise[b,k] like jax.lax.top_k output
  if (tid < TOPK) {
    float u = noise[(size_t)b * TOPK + tid];
    float g = -logf(-logf(u + 1e-10f) + 1e-10f);
    ex[tid] = (sv[tid] + g) * 10.0f;           // / TEMPERATURE(0.1)
  }
  __syncthreads();
  if (tid == 0) {
    float m = -3.4e38f;
    for (int k = 0; k < TOPK; ++k) m = fmaxf(m, ex[k]);
    float s = 0.f;
    for (int k = 0; k < TOPK; ++k) { float e = expf(ex[k] - m); ex[k] = e; s += e; }
    rdenom = 1.0f / s;
  }
  __syncthreads();
  if (tid < TOPK) {
    attn[(size_t)b * TOPK + tid]   = ex[tid] * rdenom;
    topidx[(size_t)b * TOPK + tid] = si[tid];
  }
}

// ---------------- kernel 6: gather es_db rows + weighted combine -----------
__global__ void combine_kernel(const float* __restrict__ attn,
                               const int* __restrict__ topidx,
                               const float* __restrict__ es,
                               float* __restrict__ out) {
  __shared__ float w[TOPK];
  __shared__ int   ix[TOPK];
  int b = blockIdx.x, tid = threadIdx.x;
  if (tid < TOPK) { w[tid] = attn[(size_t)b * TOPK + tid]; ix[tid] = topidx[(size_t)b * TOPK + tid]; }
  __syncthreads();
  int d = tid * 4;
  float a0 = 0.f, a1 = 0.f, a2 = 0.f, a3 = 0.f;
  for (int k = 0; k < TOPK; ++k) {
    f32x4 e = *(const f32x4*)(es + (size_t)ix[k] * D_DIM + d);
    float wk = w[k];
    a0 += wk * e[0]; a1 += wk * e[1]; a2 += wk * e[2]; a3 += wk * e[3];
  }
  f32x4 r = { a0, a1, a2, a3 };
  *(f32x4*)(out + (size_t)b * D_DIM + d) = r;
}

// ---------------- host entry ----------------
extern "C" void kernel_launch(void* const* d_in, const int* in_sizes, int n_in,
                              void* d_out, int out_size, void* d_ws, size_t ws_size,
                              hipStream_t stream) {
  (void)in_sizes; (void)n_in; (void)out_size; (void)ws_size;
  const float* query = (const float*)d_in[0];   // [1024,1024]
  const float* en_db = (const float*)d_in[1];   // [100000,1024]
  const float* es_db = (const float*)d_in[2];   // [100000,1024]
  const float* noise = (const float*)d_in[3];   // [1024,70]
  float* out = (float*)d_out;                   // [1024,1024] f32

  char* w = (char*)d_ws;
  size_t off = 0;
  auto alloc = [&](size_t bytes) -> void* {
    void* p = w + off;
    off = (off + bytes + 255) & ~(size_t)255;
    return p;
  };
  float*          qrinv  = (float*)alloc((size_t)B_Q * 4);
  float*          dbrinv = (float*)alloc((size_t)N_DB * 4);
  unsigned short* qn     = (unsigned short*)alloc((size_t)B_Q * D_DIM * 2);
  unsigned short* dbn    = (unsigned short*)alloc((size_t)N_DB * D_DIM * 2);
  float*          sims   = (float*)alloc((size_t)B_Q * NPAD * 4);
  float*          candV  = (float*)alloc((size_t)B_Q * NCH * TOPK * 4);
  int*            candI  = (int*)alloc((size_t)B_Q * NCH * TOPK * 4);
  float*          attn   = (float*)alloc((size_t)B_Q * TOPK * 4);
  int*            topidx = (int*)alloc((size_t)B_Q * TOPK * 4);

  rownorm_kernel<<<(B_Q + N_DB + 7) / 8, 256, 0, stream>>>(query, en_db, qrinv, dbrinv);
  tobf16_kernel<<<B_Q + N_DB, 256, 0, stream>>>(query, en_db, qrinv, dbrinv, qn, dbn);
  gemm_sims_kernel<<<dim3(NBLK, MBLK), 256, 0, stream>>>(qn, dbn, sims);
  topk_chunk_kernel<<<dim3(NCH, B_Q), 256, 0, stream>>>(sims, candV, candI);
  merge_softmax_kernel<<<B_Q, 256, 0, stream>>>(candV, candI, noise, attn, topidx);
  combine_kernel<<<B_Q, 256, 0, stream>>>(attn, topidx, es_db, out);
}
